// Lstm_80925773791518
// MI455X (gfx1250) — compile-verified, matrix operand order checked by reading, not run on hardware
//
#include <hip/hip_runtime.h>
#include <hip/hip_bf16.h>

// ---- types for CDNA5 WMMA ----
typedef __attribute__((ext_vector_type(16))) __bf16 v16bf;
typedef __attribute__((ext_vector_type(8)))  __bf16 v8bf;
typedef __attribute__((ext_vector_type(8)))  float  v8f;
typedef int v4i_vs __attribute__((vector_size(16)));   // matches async builtin param type

// Problem constants (from reference setup_inputs)
#define BSZ 64
#define TT  256
#define DD  1024
#define HH  1536

// GEMM tiling
#define KCH    128            // K elements staged per LDS chunk
#define LDSTR  136            // padded LDS row stride (bf16): 272B -> 68 words, 68%64=4
#define KB_PER_CH (KCH / 32)  // 4 WMMA k-blocks per chunk
#define XSTR   132            // f32 exchange row stride (words), 132%64=4

// ---- gfx1250 async global->LDS path (guarded; fallback = load + ds_store) ----
#if defined(__HIP_DEVICE_COMPILE__) && \
    __has_builtin(__builtin_amdgcn_global_load_async_to_lds_b128) && \
    __has_builtin(__builtin_amdgcn_s_wait_asynccnt)
#define ASYNC_COPY16(gp, lp)                                                   \
    __builtin_amdgcn_global_load_async_to_lds_b128(                            \
        (__attribute__((address_space(1))) v4i_vs*)(gp),                       \
        (__attribute__((address_space(3))) v4i_vs*)(lp), 0, 0)
#define ASYNC_WAIT() __builtin_amdgcn_s_wait_asynccnt(0)
#else
#define ASYNC_COPY16(gp, lp) (*(v8bf*)(lp) = *(const v8bf*)(gp))
#define ASYNC_WAIT()
#endif

// ---------------------------------------------------------------------------
// Utility kernels
// ---------------------------------------------------------------------------
__global__ void cvt_f32_to_bf16(const float* __restrict__ src,
                                __bf16* __restrict__ dst, size_t n) {
    size_t i = (size_t)blockIdx.x * blockDim.x + threadIdx.x;
    if (i < n) dst[i] = (__bf16)src[i];
}

__global__ void copy_f32(const float* __restrict__ src,
                         float* __restrict__ dst, size_t n) {
    size_t i = (size_t)blockIdx.x * blockDim.x + threadIdx.x;
    if (i < n) dst[i] = src[i];
}

// ---------------------------------------------------------------------------
// Pack fp32 row-major weight [K, N4] -> bf16 WMMA B-operand tiles.
// Tile = 32(K) x 16(N); per tile lane-major, 16 bf16 per lane:
//   lane: column n = nb*16 + (lane&15); khalf = (lane>>4)*8
//   e<8 : k = kb*32 + khalf + e ; e>=8: k = kb*32 + khalf + 8 + e
// ---------------------------------------------------------------------------
__global__ void pack_weight_bf16(const float* __restrict__ W, int K, int N4,
                                 __bf16* __restrict__ Bp, int kbOff, int nKbTot) {
    size_t idx = (size_t)blockIdx.x * blockDim.x + threadIdx.x;
    size_t total = (size_t)K * N4;
    if (idx >= total) return;
    int e    = (int)(idx & 15);
    size_t t = idx >> 4;
    int lane = (int)(t & 31);
    t >>= 5;
    int nKbL = K >> 5;
    int kb   = (int)(t % nKbL);
    int nb   = (int)(t / nKbL);
    int n = nb * 16 + (lane & 15);
    int k = kb * 32 + ((lane >> 4) * 8) + ((e < 8) ? e : (8 + e));
    __bf16 v = (__bf16)W[(size_t)k * N4 + n];
    Bp[(((size_t)nb * nKbTot + (size_t)(kbOff + kb)) * 32 + lane) * 16 + e] = v;
}

// ---------------------------------------------------------------------------
// Fused LSTM layer step:  gates = [A1|A2] @ Bp (+bias)  ->  cell update.
// Grid = Hout/32 workgroups; WG owns output columns [blk*32, blk*32+32) for
// ALL four gates: 8 waves = 4 gates x 2 n-tiles, each wave a 64x16 WMMA tile.
// A staged in double-buffered LDS chunks (async-to-LDS); B streamed with a
// 4-deep register prefetch ring. Per kk-step all 8 ds_loads are issued before
// the 4 WMMAs so the next step's LDS loads overlap current XDL execution.
// Epilogue: waves exchange f32 gate tiles via LDS (reusing the A buffers),
// then threads apply the LSTM cell and write c, h(f32), h(bf16), optional y.
// h_bf_out must differ from the A2 operand buffer (ping-pong across steps).
// ---------------------------------------------------------------------------
__global__ __launch_bounds__(256) void lstm_layer_step(
    const __bf16* __restrict__ A1, long lda1, int K1,
    const __bf16* __restrict__ A2, long lda2, int K2,
    const __bf16* __restrict__ Bp, int Hout,
    const float* __restrict__ bias,
    float* __restrict__ c_state,
    float* __restrict__ h_f32,
    __bf16* __restrict__ h_bf_out,
    float* __restrict__ y, long yStride) {
    __shared__ __attribute__((aligned(16))) __bf16 As[2][64 * LDSTR];

    const int tid  = threadIdx.x;
    const int lane = tid & 31;
    const int wave = tid >> 5;          // = gate*2 + ntile
    const int g    = wave >> 1;
    const int nt   = wave & 1;
    const int nb   = g * (Hout >> 4) + blockIdx.x * 2 + nt;
    const int col  = lane & 15;
    const int hi   = lane >> 4;
    const int koff = hi * 8;
    const int nc1  = K1 / KCH;
    const int nch  = nc1 + K2 / KCH;
    const int nKbTot = (K1 + K2) >> 5;

    // cooperative chunk loader: 64 rows x 128 bf16 (16 b128 segs/row)
    auto load_chunk = [&](int c, int buf) {
        const __bf16* base;
        long lda;
        if (c < nc1) { base = A1 + (size_t)c * KCH;          lda = lda1; }
        else         { base = A2 + (size_t)(c - nc1) * KCH;  lda = lda2; }
#pragma unroll
        for (int i = 0; i < 4; ++i) {
            int seg = tid + i * 256;
            int row = seg >> 4;
            int s   = seg & 15;
            const __bf16* gp = base + (size_t)row * lda + s * 8;
            __bf16* lp = &As[buf][row * LDSTR + s * 8];
            ASYNC_COPY16(gp, lp);
        }
    };

    v8f acc[4];
#pragma unroll
    for (int i = 0; i < 4; ++i)
        acc[i] = (v8f){0.f, 0.f, 0.f, 0.f, 0.f, 0.f, 0.f, 0.f};

    // B prefetch ring (4 deep = one chunk ahead)
    const __bf16* bp = Bp + (size_t)nb * nKbTot * 512 + (size_t)lane * 16;
    v16bf bb[KB_PER_CH];
#pragma unroll
    for (int i = 0; i < KB_PER_CH; ++i) { bb[i] = *(const v16bf*)bp; bp += 512; }

    load_chunk(0, 0);
    ASYNC_WAIT();
    __syncthreads();

    for (int c = 0; c < nch; ++c) {
        if (c + 1 < nch) load_chunk(c + 1, (c + 1) & 1);
        const __bf16* as = &As[c & 1][0];
        const __bf16* arow = as + col * LDSTR + koff;
#pragma unroll
        for (int kk = 0; kk < KB_PER_CH; ++kk) {
            v16bf bcur = bb[kk];
            bb[kk] = *(const v16bf*)bp;   // prefetch tile (c+1, kk); pad covers overrun
            bp += 512;
            // issue ALL A-fragment loads for this kk first ...
            v16bf av[4];
#pragma unroll
            for (int mt = 0; mt < 4; ++mt) {
                const __bf16* ar = arow + mt * 16 * LDSTR + kk * 32;
                v8bf alo = *(const v8bf*)ar;
                v8bf ahi = *(const v8bf*)(ar + 16);
                av[mt] = __builtin_shufflevector(alo, ahi, 0, 1, 2, 3, 4, 5, 6, 7,
                                                 8, 9, 10, 11, 12, 13, 14, 15);
            }
            // ... then the 4 WMMAs back-to-back (next kk's loads overlap these)
#pragma unroll
            for (int mt = 0; mt < 4; ++mt) {
                acc[mt] = __builtin_amdgcn_wmma_f32_16x16x32_bf16(
                    false, av[mt], false, bcur, (short)0, acc[mt], false, false);
            }
        }
        if (c + 1 < nch) {
            ASYNC_WAIT();
            __syncthreads();
        }
    }

    // ---- epilogue: exchange gate tiles through LDS, then cell update ----
    __syncthreads();                       // all waves done reading As
    float* xch = (float*)&As[0][0];        // 64 rows x 128 cols f32, stride XSTR

    // store: wave w's tile occupies columns [w*16, w*16+16)
#pragma unroll
    for (int mt = 0; mt < 4; ++mt) {
        int rbase = mt * 16 + hi * 8;
#pragma unroll
        for (int r = 0; r < 8; ++r)
            xch[(rbase + r) * XSTR + wave * 16 + col] = acc[mt][r];
    }
    __syncthreads();

    // cell: 64 rows x 32 cols = 2048 elems, 8 per thread
    const int n0 = blockIdx.x * 32;
#pragma unroll
    for (int i = 0; i < 8; ++i) {
        int p   = tid + i * 256;
        int row = p >> 5;
        int nl  = p & 31;
        int nt2 = nl >> 4;
        int cl  = nl & 15;
        float gi = xch[row * XSTR + (0 + nt2) * 16 + cl];
        float gf = xch[row * XSTR + (2 + nt2) * 16 + cl];
        float gg = xch[row * XSTR + (4 + nt2) * 16 + cl];
        float go = xch[row * XSTR + (6 + nt2) * 16 + cl];
        int n = n0 + nl;
        gi += bias[n];
        gf += bias[Hout + n];
        gg += bias[2 * Hout + n];
        go += bias[3 * Hout + n];
        float si = 1.f / (1.f + __expf(-gi));
        float sf = 1.f / (1.f + __expf(-gf));
        float so = 1.f / (1.f + __expf(-go));
        size_t ci = (size_t)row * Hout + n;
        float c = sf * c_state[ci] + si * tanhf(gg);
        float h = so * tanhf(c);
        c_state[ci]  = c;
        h_f32[ci]    = h;
        h_bf_out[ci] = (__bf16)h;
        if (y) y[(size_t)row * yStride + n] = h;
    }
}

// ---------------------------------------------------------------------------
// Launch
// ---------------------------------------------------------------------------
extern "C" void kernel_launch(void* const* d_in, const int* in_sizes, int n_in,
                              void* d_out, int out_size, void* d_ws, size_t ws_size,
                              hipStream_t stream) {
    (void)in_sizes; (void)n_in; (void)out_size; (void)ws_size;

    const float* x     = (const float*)d_in[0];   // [B, T, D]
    const float* h0_0  = (const float*)d_in[1];   // [B, H]
    const float* c0_0  = (const float*)d_in[2];   // [B, H]
    const float* h0_1  = (const float*)d_in[3];   // [B, D]
    const float* c0_1  = (const float*)d_in[4];   // [B, D]
    const float* W_ih0 = (const float*)d_in[5];   // [D, 4H]
    const float* W_hh0 = (const float*)d_in[6];   // [H, 4H]
    const float* b0    = (const float*)d_in[7];   // [4H]
    const float* W_ih1 = (const float*)d_in[8];   // [H, 4D]
    const float* W_hh1 = (const float*)d_in[9];   // [D, 4D]
    const float* b1    = (const float*)d_in[10];  // [4D]
    float* out = (float*)d_out;                   // outputs [B,T,D] then h0,c0,h1,c1

    const int N4_0 = 4 * HH;            // 6144
    const int N4_1 = 4 * DD;            // 4096
    const int nKbTot = (DD + HH) >> 5;  // 80 K-blocks

    // ---- workspace carve-up (256B aligned; packed buffers get 8KB tail pad
    //      so the 4-deep B prefetch overrun stays in-bounds) ----
    auto align_up = [](size_t v) { return (v + 255) & ~(size_t)255; };
    char* w = (char*)d_ws;
    size_t off = 0;
    __bf16* xb  = (__bf16*)(w + off); off = align_up(off + (size_t)BSZ * TT * DD * 2);
    __bf16* Wp0 = (__bf16*)(w + off); off = align_up(off + (size_t)(DD + HH) * N4_0 * 2 + 8192);
    __bf16* Wp1 = (__bf16*)(w + off); off = align_up(off + (size_t)(DD + HH) * N4_1 * 2 + 8192);
    float*  h0f = (float*) (w + off); off = align_up(off + (size_t)BSZ * HH * 4);
    float*  c0s = (float*) (w + off); off = align_up(off + (size_t)BSZ * HH * 4);
    float*  h1f = (float*) (w + off); off = align_up(off + (size_t)BSZ * DD * 4);
    float*  c1s = (float*) (w + off); off = align_up(off + (size_t)BSZ * DD * 4);
    __bf16* h0b[2];
    h0b[0] = (__bf16*)(w + off); off = align_up(off + (size_t)BSZ * HH * 2);
    h0b[1] = (__bf16*)(w + off); off = align_up(off + (size_t)BSZ * HH * 2);
    __bf16* h1b[2];
    h1b[0] = (__bf16*)(w + off); off = align_up(off + (size_t)BSZ * DD * 2);
    h1b[1] = (__bf16*)(w + off); off = align_up(off + (size_t)BSZ * DD * 2);

    const int TPB = 256;
    auto blocks = [](size_t n) { return (unsigned)((n + 255) / 256); };

    // ---- one-time per launch: convert / pack / init state ----
    {
        size_t nx = (size_t)BSZ * TT * DD;
        cvt_f32_to_bf16<<<blocks(nx), TPB, 0, stream>>>(x, xb, nx);

        size_t n;
        n = (size_t)DD * N4_0;
        pack_weight_bf16<<<blocks(n), TPB, 0, stream>>>(W_ih0, DD, N4_0, Wp0, 0, nKbTot);
        n = (size_t)HH * N4_0;
        pack_weight_bf16<<<blocks(n), TPB, 0, stream>>>(W_hh0, HH, N4_0, Wp0, DD >> 5, nKbTot);
        n = (size_t)HH * N4_1;
        pack_weight_bf16<<<blocks(n), TPB, 0, stream>>>(W_ih1, HH, N4_1, Wp1, 0, nKbTot);
        n = (size_t)DD * N4_1;
        pack_weight_bf16<<<blocks(n), TPB, 0, stream>>>(W_hh1, DD, N4_1, Wp1, HH >> 5, nKbTot);

        size_t nh0 = (size_t)BSZ * HH, nh1 = (size_t)BSZ * DD;
        cvt_f32_to_bf16<<<blocks(nh0), TPB, 0, stream>>>(h0_0, h0b[0], nh0);
        cvt_f32_to_bf16<<<blocks(nh1), TPB, 0, stream>>>(h0_1, h1b[0], nh1);
        copy_f32<<<blocks(nh0), TPB, 0, stream>>>(h0_0, h0f, nh0);
        copy_f32<<<blocks(nh0), TPB, 0, stream>>>(c0_0, c0s, nh0);
        copy_f32<<<blocks(nh1), TPB, 0, stream>>>(h0_1, h1f, nh1);
        copy_f32<<<blocks(nh1), TPB, 0, stream>>>(c0_1, c1s, nh1);
    }

    // ---- sequential time loop: 2 fused kernels per step ----
    for (int t = 0; t < TT; ++t) {
        int cur = t & 1, nxt = (t + 1) & 1;
        // layer 0: A = [x_t | h0_prev] -> h0_new, c0
        lstm_layer_step<<<HH / 32, TPB, 0, stream>>>(
            xb + (size_t)t * DD, (long)TT * DD, DD,
            h0b[cur], (long)HH, HH,
            Wp0, HH, b0,
            c0s, h0f, h0b[nxt],
            (float*)nullptr, 0);
        // layer 1: A = [h0_new | h1_prev] -> h1_new, c1, y[:, t, :]
        lstm_layer_step<<<DD / 32, TPB, 0, stream>>>(
            h0b[nxt], (long)HH, HH,
            h1b[cur], (long)DD, DD,
            Wp1, DD, b1,
            c1s, h1f, h1b[nxt],
            out + (size_t)t * DD, (long)TT * DD);
    }

    // ---- final states -> tail of d_out: h0, c0, h1, c1 ----
    size_t o = (size_t)BSZ * TT * DD;
    size_t nh0 = (size_t)BSZ * HH, nh1 = (size_t)BSZ * DD;
    copy_f32<<<blocks(nh0), TPB, 0, stream>>>(h0f, out + o, nh0); o += nh0;
    copy_f32<<<blocks(nh0), TPB, 0, stream>>>(c0s, out + o, nh0); o += nh0;
    copy_f32<<<blocks(nh1), TPB, 0, stream>>>(h1f, out + o, nh1); o += nh1;
    copy_f32<<<blocks(nh1), TPB, 0, stream>>>(c1s, out + o, nh1);
}
